// Top1Gate_21655225107172
// MI455X (gfx1250) — matrix operand order repaired and checked by reading, not compiled, not run on hardware
//
#include <hip/hip_runtime.h>
#include <hip/hip_bf16.h>

#define N_TOKENS    16384
#define MODEL_DIM   4096
#define NUM_EXPERTS 128
#define TEMP_L_AUX  0.07f
#define COS_EPS     1e-4f
#define MTILE       128
#define KC          64
#define LDA         72   // A tile stride: 64+8 pad, rows 144B (16B-aligned), conflict-free
#define LDB         72   // B tile stride: 64+8 pad, rows 144B (16B-aligned), conflict-free

typedef __attribute__((ext_vector_type(16))) __bf16 v16bf;
typedef __attribute__((ext_vector_type(8)))  float  v8f;

union FragBF { v16bf v; uint4 q[2]; };

__device__ __forceinline__ unsigned short f2bf(float f) {
  union { float f; unsigned u; } x; x.f = f;
  unsigned u = x.u + 0x7FFFu + ((x.u >> 16) & 1u);   // round-to-nearest-even
  return (unsigned short)(u >> 16);
}

// packed f32x2 -> bf16x2 (hardware cvt_pk where available)
__device__ __forceinline__ void store_bf2(unsigned short* dst, float a, float b) {
  __hip_bfloat162 p = __float22bfloat162_rn(make_float2(a, b));
  *reinterpret_cast<__hip_bfloat162*>(dst) = p;
}

__device__ __forceinline__ void loadB(FragBF& f, const unsigned short* base) {
  f.q[0] = *(const uint4*)(base);
  f.q[1] = *(const uint4*)(base + 16);
}

// ---------------- kernel 0: zero the expert counters --------------------
__global__ void k0_zero(int* __restrict__ counts) {
  int t = threadIdx.x;
  if (t < NUM_EXPERTS) counts[t] = 0;
}

// ---------------- kernel 1: L2-normalize wg rows -> bf16 ----------------
__global__ __launch_bounds__(256) void k1_norm_wg(const float* __restrict__ wg,
                                                  unsigned short* __restrict__ wgn) {
  __shared__ float red[256];
  int e = blockIdx.x, t = threadIdx.x;
  const float* row = wg + (size_t)e * MODEL_DIM;
  float vals[16]; float ss = 0.f;
#pragma unroll
  for (int j = 0; j < 16; ++j) { float v = row[j * 256 + t]; vals[j] = v; ss += v * v; }
  red[t] = ss; __syncthreads();
  if (t == 0) { float s = 0.f; for (int i = 0; i < 256; ++i) s += red[i]; red[0] = s; }
  __syncthreads();
  float rinv = 1.0f / fmaxf(sqrtf(red[0]), COS_EPS);
  unsigned short* orow = wgn + (size_t)e * MODEL_DIM;
#pragma unroll
  for (int j = 0; j < 16; ++j) orow[j * 256 + t] = f2bf(vals[j] * rinv);
}

// ------- kernel 2: fused GEMM (WMMA bf16) + row-norm + argmax + gates ----
__global__ __launch_bounds__(256) void k2_gemm_route(
    const float* __restrict__ x, const unsigned short* __restrict__ wgn,
    int* __restrict__ tte, int* __restrict__ gcounts,
    float* __restrict__ me_partial) {
  __shared__ unsigned short ldsA[MTILE * LDA];         // 18 KB bf16 x-tile
  __shared__ unsigned short ldsB[NUM_EXPERTS * LDB];   // 18 KB bf16 wg-slab
  __shared__ float rowsq[MTILE];
  __shared__ float meW[8 * NUM_EXPERTS];
  __shared__ int   cntL[NUM_EXPERTS];

  const int t = threadIdx.x;
  const int w = t >> 5, lane = t & 31;
  const int half = lane >> 4, ln = lane & 15;
  const int tokenBase = blockIdx.x * MTILE;
  const int col4 = (t & 15) * 4;          // fp32 column group this thread loads (A)
  const int rbase = t >> 4;               // base row (t/16), rows rbase+16*i (A)
  const int be = t >> 3;                  // B staging: 8 threads per expert row
  const int bq = (t & 7) * 8;             // B staging: ushort column (uint4 granules)

  if (t < MTILE) { rowsq[t] = 0.f; }
  if (t < NUM_EXPERTS) { cntL[t] = 0; }
  __syncthreads();

  v8f acc[8] = {};                         // 8 N-tiles of 16x16 f32
  float ssq[8];
#pragma unroll
  for (int i = 0; i < 8; ++i) ssq[i] = 0.f;

  // per-lane base pointers for fragment loads (invariant across chunks)
  const unsigned short* aBase = &ldsA[(w * 16 + ln) * LDA + half * 8];
  const unsigned short* bBase = &ldsB[ln * LDB + half * 8];

  for (int c = 0; c < MODEL_DIM / KC; ++c) {
    const int k0 = c * KC;
    // --- stage B slab: wgn[0..127][k0..k0+64) -> LDS, each byte once/block
#pragma unroll
    for (int i = 0; i < 4; ++i) {
      const int e = be + 32 * i;           // 8 threads x 8 uint4 = one 64-col row
      const uint4 bv = *(const uint4*)(wgn + (size_t)e * MODEL_DIM + k0 + bq);
      *(uint4*)(&ldsB[e * LDB + bq]) = bv;
    }
    // --- stage A tile: coalesced float4 loads, sum-of-squares, bf16 -> LDS
#pragma unroll
    for (int i = 0; i < 8; ++i) {
      const int row = rbase + 16 * i;
      const float4 v = *(const float4*)(x + (size_t)(tokenBase + row) * MODEL_DIM + k0 + col4);
      ssq[i] += v.x * v.x + v.y * v.y + v.z * v.z + v.w * v.w;
      unsigned short* dst = &ldsA[row * LDA + col4];
      store_bf2(dst + 0, v.x, v.y);
      store_bf2(dst + 2, v.z, v.w);
      if (c + 1 < MODEL_DIM / KC)
        __builtin_prefetch(x + (size_t)(tokenBase + row) * MODEL_DIM + k0 + KC + col4, 0, 1);
    }
    __syncthreads();

    // fence: nothing from the staging phase may drift into the WMMA burst
    __builtin_amdgcn_sched_barrier(0);

    // --- both A frags for this chunk (kk=0 and kk=32), loaded once
    FragBF a0, a1;
    a0.q[0] = *(const uint4*)(aBase);        a0.q[1] = *(const uint4*)(aBase + 16);
    a1.q[0] = *(const uint4*)(aBase + 32);   a1.q[1] = *(const uint4*)(aBase + 48);

    // --- 16 WMMA steps, B frags pipelined 2 deep through a 3-slot rotator.
    // step s: n = s>>1 (N tile), kk = (s&1)*32.
    FragBF b[3];
    loadB(b[0], bBase + 0);                  // (n=0, kk=0)
    loadB(b[1], bBase + 32);                 // (n=0, kk=32)
#pragma unroll
    for (int s = 0; s < 16; ++s) {
      if (s + 2 < 16) {
        const int s2 = s + 2;
        loadB(b[s2 % 3], bBase + (s2 >> 1) * (16 * LDB) + (s2 & 1) * 32);
      }
      const FragBF& a = (s & 1) ? a1 : a0;
      acc[s >> 1] = __builtin_amdgcn_wmma_f32_16x16x32_bf16(
          false, a.v, false, b[s % 3].v, (short)0, acc[s >> 1], false, false);
    }

    // Pin the software pipeline (mask 0x100 = DS read, 0x008 = MFMA/WMMA):
    //   [8 ds_read] (a0,a1,b0,b1) ; 14 x [1 wmma, 2 ds_read] ; [2 wmma]
    __builtin_amdgcn_sched_group_barrier(0x100, 8, 0);
#pragma unroll
    for (int s = 0; s < 14; ++s) {
      __builtin_amdgcn_sched_group_barrier(0x008, 1, 0);
      __builtin_amdgcn_sched_group_barrier(0x100, 2, 0);
    }
    __builtin_amdgcn_sched_group_barrier(0x008, 2, 0);
    __builtin_amdgcn_sched_barrier(0);

    __syncthreads();
  }

#pragma unroll
  for (int i = 0; i < 8; ++i) atomicAdd(&rowsq[rbase + 16 * i], ssq[i]);
  __syncthreads();

  // Epilogue: logits = acc / max(||x_row||, eps); per-row argmax + softmax
  const float invT = 1.0f / TEMP_L_AUX;
  float gacc[8];
#pragma unroll
  for (int n = 0; n < 8; ++n) gacc[n] = 0.f;

  for (int j = 0; j < 8; ++j) {
    const int rowA = w * 16 + j + half * 8;   // row this lane's element j belongs to
    const float rinv = 1.0f / fmaxf(sqrtf(rowsq[rowA]), COS_EPS);
    float v[8];
#pragma unroll
    for (int n = 0; n < 8; ++n) v[n] = acc[n][j] * rinv;

    // local argmax (ascending n keeps lowest expert index on ties)
    float bm = v[0]; int bi = ln;
#pragma unroll
    for (int n = 1; n < 8; ++n)
      if (v[n] > bm) { bm = v[n]; bi = 16 * n + ln; }
    // butterfly within the 16-lane half (wave32)
#pragma unroll
    for (int m = 1; m <= 8; m <<= 1) {
      float ov = __shfl_xor(bm, m, 32);
      int   oi = __shfl_xor(bi, m, 32);
      if (ov > bm || (ov == bm && oi < bi)) { bm = ov; bi = oi; }
    }
    // stable softmax(logits / TEMP_L_AUX); bm == row max after butterfly
    float ex[8]; float s = 0.f;
#pragma unroll
    for (int n = 0; n < 8; ++n) { ex[n] = __expf((v[n] - bm) * invT); s += ex[n]; }
#pragma unroll
    for (int m = 1; m <= 8; m <<= 1) s += __shfl_xor(s, m, 32);
    const float sinv = 1.0f / s;
#pragma unroll
    for (int n = 0; n < 8; ++n) gacc[n] += ex[n] * sinv;

    if (ln == 0) {
      tte[tokenBase + rowA] = bi;
      atomicAdd(&cntL[bi], 1);
    }
  }
  // combine the two halves deterministically, then fixed-order wave reduce
#pragma unroll
  for (int n = 0; n < 8; ++n) gacc[n] += __shfl_xor(gacc[n], 16, 32);
  if (half == 0) {
#pragma unroll
    for (int n = 0; n < 8; ++n) meW[w * NUM_EXPERTS + 16 * n + ln] = gacc[n];
  }
  __syncthreads();
  if (t < NUM_EXPERTS) {
    float s = 0.f;
    for (int ww = 0; ww < 8; ++ww) s += meW[ww * NUM_EXPERTS + t];
    me_partial[blockIdx.x * NUM_EXPERTS + t] = s;   // deterministic per-block sum
    if (cntL[t]) atomicAdd(&gcounts[t], cntL[t]);   // integer: order-independent
  }
}

// ---- kernel 3: stable counting sort + l_aux + splits (single block) -----
__global__ __launch_bounds__(256) void k3_finalize(
    const int* __restrict__ gcounts, const float* __restrict__ me_partial,
    const int* __restrict__ tte, float* __restrict__ out) {
  __shared__ unsigned char  t2e[N_TOKENS];              // 16 KB
  __shared__ unsigned short H[128 * NUM_EXPERTS];       // 32 KB chunk histograms
  __shared__ int   cnts[NUM_EXPERTS];
  __shared__ int   offs[NUM_EXPERTS];
  __shared__ float sme[NUM_EXPERTS];
  const int t = threadIdx.x;

  if (t < NUM_EXPERTS) {
    cnts[t] = gcounts[t];
    float s = 0.f;
    for (int b = 0; b < N_TOKENS / MTILE; ++b) s += me_partial[b * NUM_EXPERTS + t];
    sme[t] = s;
  }
  for (int i = t; i < N_TOKENS; i += 256) t2e[i] = (unsigned char)tte[i];
  for (int i = t; i < 128 * NUM_EXPERTS; i += 256) H[i] = 0;
  __syncthreads();

  // per-chunk histograms (chunk = 128 consecutive tokens, one thread each)
  if (t < 128) {
    const int base = t * 128;
    unsigned short* h = &H[t * NUM_EXPERTS];
    for (int i = 0; i < 128; ++i) h[t2e[base + i]]++;
  }
  __syncthreads();
  // per-expert exclusive prefix across chunks (stability across chunks)
  if (t < NUM_EXPERTS) {
    unsigned short run = 0;
    for (int c = 0; c < 128; ++c) {
      unsigned short tmp = H[c * NUM_EXPERTS + t];
      H[c * NUM_EXPERTS + t] = run;
      run += tmp;
    }
  }
  __syncthreads();
  if (t == 0) {
    int s = 0;
    for (int e = 0; e < NUM_EXPERTS; ++e) { offs[e] = s; s += cnts[e]; }
    // l_aux = (me . ce) * E, ce = counts/N + 1e-6
    float la = 0.f;
    for (int e = 0; e < NUM_EXPERTS; ++e)
      la += sme[e] * ((float)cnts[e] * (1.0f / (float)N_TOKENS) + 1e-6f);
    out[0] = la * (float)NUM_EXPERTS;
  }
  __syncthreads();
  // stable emit: pos = expert base + chunk-exclusive offset + local occurrence
  if (t < 128) {
    const int base = t * 128;
    unsigned short* h = &H[t * NUM_EXPERTS];
    for (int i = 0; i < 128; ++i) {
      const int e = t2e[base + i];
      const int pos = offs[e] + h[e];
      h[e]++;
      out[1 + pos] = (float)(base + i);
    }
  }
  if (t < NUM_EXPERTS) {
    out[1 + N_TOKENS + t]               = (float)cnts[t];   // input_splits
    out[1 + N_TOKENS + NUM_EXPERTS + t] = (float)cnts[t];   // output_splits
  }
}

extern "C" void kernel_launch(void* const* d_in, const int* in_sizes, int n_in,
                              void* d_out, int out_size, void* d_ws, size_t ws_size,
                              hipStream_t stream) {
  const float* x  = (const float*)d_in[0];
  const float* wg = (const float*)d_in[1];
  // d_in[2] (gating_t) does not affect any output: argmax is invariant under
  // the monotone sigmoid/temperature, and l_aux uses the fixed TEMP_L_AUX.
  (void)in_sizes; (void)n_in; (void)out_size; (void)ws_size;

  char* ws = (char*)d_ws;
  unsigned short* wgn = (unsigned short*)ws;                               // 1 MB bf16
  int*   tte    = (int*)(ws + (size_t)NUM_EXPERTS * MODEL_DIM * 2);        // 64 KB
  int*   counts = (int*)((char*)tte + (size_t)N_TOKENS * 4);               // 512 B
  float* mep    = (float*)((char*)counts + 512);                           // 64 KB

  k0_zero<<<1, 256, 0, stream>>>(counts);
  k1_norm_wg<<<NUM_EXPERTS, 256, 0, stream>>>(wg, wgn);
  k2_gemm_route<<<N_TOKENS / MTILE, 256, 0, stream>>>(x, wgn, tte, counts, mep);
  k3_finalize<<<1, 256, 0, stream>>>(counts, mep, tte, (float*)d_out);
}
